// MHAandRoPE_57105885167819
// MI455X (gfx1250) — matrix operand order repaired and checked by reading, not compiled, not run on hardware
//
#include <hip/hip_runtime.h>

#define B_ 2
#define S_ 2048
#define D_ 1024
#define H_ 16
#define HD_ 64
#define HALF_ 32

typedef __bf16 bf16_t;
typedef bf16_t v16bf __attribute__((ext_vector_type(16)));
typedef bf16_t v8bf  __attribute__((ext_vector_type(8)));
typedef float  v8f   __attribute__((ext_vector_type(8)));

union V16U { v16bf v; v8bf h[2]; };

static __device__ inline v8f wmma_bf16(v16bf a, v16bf b, v8f c) {
  return __builtin_amdgcn_wmma_f32_16x16x32_bf16(
      /*neg_a=*/false, a, /*neg_b=*/false, b,
      /*c_mod=*/(short)0, c, /*reuse_a=*/false, /*reuse_b=*/false);
}

// Build a 16-bit A-operand (16x32) from a row pointer:
// lanes 0-15 take K = {0..15}, lanes 16-31 take K = {16..31}.
static __device__ inline v16bf make_a(const bf16_t* base, int hsel) {
  V16U u;
  u.v = *(const v16bf*)(base + hsel * 16);
  return u.v;
}

// ---------------- prep kernels ----------------

__global__ void k_cvt_x(const float* __restrict__ x, bf16_t* __restrict__ xb, int n) {
  int i = blockIdx.x * blockDim.x + threadIdx.x;
  if (i < n) xb[i] = (bf16_t)x[i];
}

__global__ void k_cvt_w(const float* __restrict__ wq, const float* __restrict__ wk,
                        const float* __restrict__ wv, const float* __restrict__ wo,
                        bf16_t* __restrict__ tq, bf16_t* __restrict__ tk,
                        bf16_t* __restrict__ tv, bf16_t* __restrict__ to_) {
  int i = blockIdx.x * blockDim.x + threadIdx.x;        // over D*D
  int m = blockIdx.y;
  const float* src = (m == 0) ? wq : (m == 1) ? wk : (m == 2) ? wv : wo;
  bf16_t* dst      = (m == 0) ? tq : (m == 1) ? tk : (m == 2) ? tv : to_;
  int k = i / D_;
  int n = i % D_;
  dst[(size_t)n * D_ + k] = (bf16_t)src[(size_t)k * D_ + n];  // transposed: Wt[n][k]
}

// ---------------- bf16 WMMA GEMM core ----------------
// One wave computes a 32x64 tile: 2 row sub-tiles x 4 col sub-tiles,
// software-pipelined so iteration k+1's loads overlap iteration k's WMMAs.

template <typename OutT>
static __device__ inline void gemm_tile_32x64(const bf16_t* __restrict__ a_,
                                              const bf16_t* __restrict__ wt,
                                              OutT* __restrict__ out,
                                              int r0, int n0) {
  const int lane = threadIdx.x;
  const int mrow = lane & 15;
  const int hsel = lane >> 4;

  const bf16_t* arow0 = a_ + (size_t)(r0 + mrow) * D_;
  const bf16_t* arow1 = arow0 + (size_t)16 * D_;

  v8f acc[8] = {};

  // prologue: k-step 0 operands
  v16bf a0 = make_a(arow0, hsel);
  v16bf a1 = make_a(arow1, hsel);
  v16bf bb[4];
#pragma unroll
  for (int t = 0; t < 4; ++t)
    bb[t] = *(const v16bf*)(wt + (size_t)(n0 + t * 16 + mrow) * D_ + hsel * 16);

#pragma unroll 2
  for (int kb = 0; kb < D_ / 32 - 1; ++kb) {
    const int kn = (kb + 1) * 32;
    __builtin_prefetch(arow0 + kn + 128, 0, 1);
    // issue next iteration's loads first (partial loadcnt waits on WMMAs below)
    v16bf na0 = make_a(arow0 + kn, hsel);
    v16bf na1 = make_a(arow1 + kn, hsel);
    v16bf nb[4];
#pragma unroll
    for (int t = 0; t < 4; ++t)
      nb[t] = *(const v16bf*)(wt + (size_t)(n0 + t * 16 + mrow) * D_ + kn + hsel * 16);
#pragma unroll
    for (int t = 0; t < 4; ++t) {
      acc[t]     = wmma_bf16(a0, bb[t], acc[t]);
      acc[4 + t] = wmma_bf16(a1, bb[t], acc[4 + t]);
    }
    a0 = na0; a1 = na1;
#pragma unroll
    for (int t = 0; t < 4; ++t) bb[t] = nb[t];
  }
  // epilogue k-step
#pragma unroll
  for (int t = 0; t < 4; ++t) {
    acc[t]     = wmma_bf16(a0, bb[t], acc[t]);
    acc[4 + t] = wmma_bf16(a1, bb[t], acc[4 + t]);
  }

#pragma unroll
  for (int mt = 0; mt < 2; ++mt)
#pragma unroll
    for (int t = 0; t < 4; ++t)
#pragma unroll
      for (int v = 0; v < 8; ++v) {
        int row = r0 + mt * 16 + v + 8 * hsel;
        int col = n0 + t * 16 + mrow;
        out[(size_t)row * D_ + col] = (OutT)acc[mt * 4 + t][v];
      }
}

__global__ void k_gemm_qkv(const bf16_t* __restrict__ xb,
                           const bf16_t* __restrict__ wtq,
                           const bf16_t* __restrict__ wtk,
                           const bf16_t* __restrict__ wtv,
                           bf16_t* __restrict__ qo,
                           bf16_t* __restrict__ ko,
                           bf16_t* __restrict__ vo) {
  const bf16_t* wt = (blockIdx.z == 0) ? wtq : (blockIdx.z == 1) ? wtk : wtv;
  bf16_t* out      = (blockIdx.z == 0) ? qo  : (blockIdx.z == 1) ? ko  : vo;
  gemm_tile_32x64<bf16_t>(xb, wt, out, blockIdx.x * 32, blockIdx.y * 64);
}

__global__ void k_gemm_out(const bf16_t* __restrict__ a_, const bf16_t* __restrict__ wt,
                           float* __restrict__ out) {
  gemm_tile_32x64<float>(a_, wt, out, blockIdx.x * 32, blockIdx.y * 64);
}

// ---------------- RoPE + head repack + V transpose ----------------
// qr/kr: [b][h][s][hd] bf16 ; vt: [b][h][hd][s] bf16

__global__ void k_rope_pack(const bf16_t* __restrict__ qt, const bf16_t* __restrict__ kt,
                            const bf16_t* __restrict__ vin,
                            const float* __restrict__ cosf_, const float* __restrict__ sinf_,
                            bf16_t* __restrict__ qr, bf16_t* __restrict__ kr,
                            bf16_t* __restrict__ vt) {
  int idx = blockIdx.x * blockDim.x + threadIdx.x;   // over B*S*H*HALF = 2M
  if (idx >= B_ * S_ * H_ * HALF_) return;
  int i = idx & (HALF_ - 1);
  int h = (idx >> 5) & (H_ - 1);
  int s = (idx >> 9) & (S_ - 1);
  int b = idx >> 20;
  float c  = cosf_[s * HALF_ + i];
  float sn = sinf_[s * HALF_ + i];
  size_t src = ((size_t)(b * S_ + s)) * D_ + h * HD_ + 2 * i;
  float q0 = (float)qt[src], q1 = (float)qt[src + 1];
  float k0 = (float)kt[src], k1 = (float)kt[src + 1];
  size_t dst = (((size_t)(b * H_ + h)) * S_ + s) * HD_ + 2 * i;
  qr[dst]     = (bf16_t)(q0 * c - q1 * sn);
  qr[dst + 1] = (bf16_t)(q0 * sn + q1 * c);
  kr[dst]     = (bf16_t)(k0 * c - k1 * sn);
  kr[dst + 1] = (bf16_t)(k0 * sn + k1 * c);
  size_t vdst = (((size_t)(b * H_ + h)) * HD_ + 2 * i) * S_ + s;
  vt[vdst]      = vin[src];
  vt[vdst + S_] = vin[src + 1];
}

// ---------------- causal flash attention (one wave per 16 q-rows) ----------------

__global__ void k_attn(const bf16_t* __restrict__ qr, const bf16_t* __restrict__ kr,
                       const bf16_t* __restrict__ vt, bf16_t* __restrict__ attn) {
  __shared__ bf16_t pbuf[16 * 32];
  const int lane = threadIdx.x;
  const int mrow = lane & 15;
  const int hsel = lane >> 4;
  const int qt = blockIdx.x;                 // 0..S/16-1
  const int h = blockIdx.y, b = blockIdx.z;
  const int q0 = qt * 16;
  const size_t bh = (size_t)(b * H_ + h);

  // Q tile A-operands, K-dim 64 split in two 16x32 operands
  const bf16_t* qbase = qr + (bh * S_ + q0 + mrow) * HD_;
  v16bf qa0 = make_a(qbase, hsel);
  v16bf qa1 = make_a(qbase + 32, hsel);

  v8f acc[4] = {};
  float m_[8], l_[8];
#pragma unroll
  for (int v = 0; v < 8; ++v) { m_[v] = -__builtin_inff(); l_[v] = 0.0f; }

  const int nkb = (qt >> 1) + 1;             // causal: key blocks of 32
  for (int kb = 0; kb < nkb; ++kb) {
    const int k0 = kb * 32;

    // scores: two 16x16 C tiles covering keys k0..k0+31, K-dim 64
    v8f s0 = {}, s1 = {};
    {
      const bf16_t* kp = kr + (bh * S_ + k0 + mrow) * HD_ + hsel * 16;
      v16bf b0 = *(const v16bf*)kp;
      v16bf b1 = *(const v16bf*)(kp + 32);
      s0 = wmma_bf16(qa0, b0, s0);
      s0 = wmma_bf16(qa1, b1, s0);
      const bf16_t* kp2 = kp + 16 * HD_;
      v16bf b2 = *(const v16bf*)kp2;
      v16bf b3 = *(const v16bf*)(kp2 + 32);
      s1 = wmma_bf16(qa0, b2, s1);
      s1 = wmma_bf16(qa1, b3, s1);
    }

    // online softmax; C layout row = v + 8*hsel, col = mrow (per half-wave)
#pragma unroll
    for (int v = 0; v < 8; ++v) {
      int row = v + 8 * hsel;
      int qabs = q0 + row;
      float sv0 = s0[v] * 0.125f;            // 1/sqrt(64)
      float sv1 = s1[v] * 0.125f;
      if (k0 + mrow > qabs)      sv0 = -__builtin_inff();
      if (k0 + 16 + mrow > qabs) sv1 = -__builtin_inff();
      float rm = fmaxf(sv0, sv1);
#pragma unroll
      for (int o = 1; o < 16; o <<= 1) rm = fmaxf(rm, __shfl_xor(rm, o, 32));
      float nm = fmaxf(m_[v], rm);
      float alpha = __expf(m_[v] - nm);
      float p0 = __expf(sv0 - nm);
      float p1 = __expf(sv1 - nm);
      float rs = p0 + p1;
#pragma unroll
      for (int o = 1; o < 16; o <<= 1) rs += __shfl_xor(rs, o, 32);
      l_[v] = l_[v] * alpha + rs;
      m_[v] = nm;
#pragma unroll
      for (int t = 0; t < 4; ++t) acc[t][v] *= alpha;
      pbuf[row * 32 + mrow]      = (bf16_t)p0;   // P tile, row-major 16x32
      pbuf[row * 32 + 16 + mrow] = (bf16_t)p1;
    }
    __syncthreads();

    // P (16x32) as A-operand, V tiles as B-operand from Vt[b][h][hd][s]
    v16bf pa = make_a(&pbuf[mrow * 32], hsel);
#pragma unroll
    for (int t = 0; t < 4; ++t) {
      const bf16_t* vp = vt + (bh * HD_ + t * 16 + mrow) * S_ + k0 + hsel * 16;
      v16bf vb = *(const v16bf*)vp;
      acc[t] = wmma_bf16(pa, vb, acc[t]);
    }
    __syncthreads();
  }

  // normalize and store attn output as bf16 [b][s][D]
#pragma unroll
  for (int v = 0; v < 8; ++v) {
    int row = v + 8 * hsel;
    float inv = 1.0f / l_[v];
#pragma unroll
    for (int t = 0; t < 4; ++t) {
      attn[((size_t)(b * S_ + q0 + row)) * D_ + h * HD_ + t * 16 + mrow] =
          (bf16_t)(acc[t][v] * inv);
    }
  }
}

// ---------------- launch ----------------

extern "C" void kernel_launch(void* const* d_in, const int* in_sizes, int n_in,
                              void* d_out, int out_size, void* d_ws, size_t ws_size,
                              hipStream_t stream) {
  const float* x    = (const float*)d_in[0];
  const float* fcos = (const float*)d_in[1];
  const float* fsin = (const float*)d_in[2];
  const float* Wq   = (const float*)d_in[3];
  const float* Wk   = (const float*)d_in[4];
  const float* Wv   = (const float*)d_in[5];
  const float* Wout = (const float*)d_in[6];
  float* out = (float*)d_out;

  char* ws = (char*)d_ws;
  const size_t MB = 1024u * 1024u;
  bf16_t* xb   = (bf16_t*)(ws + 0);          //  8 MB: x as bf16
  bf16_t* wtq  = (bf16_t*)(ws + 8 * MB);     //  2 MB each: transposed bf16 weights
  bf16_t* wtk  = (bf16_t*)(ws + 10 * MB);
  bf16_t* wtv  = (bf16_t*)(ws + 12 * MB);
  bf16_t* wto  = (bf16_t*)(ws + 14 * MB);
  bf16_t* qtmp = (bf16_t*)(ws + 16 * MB);    //  8 MB each: Q/K/V pre-RoPE [b][s][D]
  bf16_t* ktmp = (bf16_t*)(ws + 24 * MB);
  bf16_t* vtmp = (bf16_t*)(ws + 32 * MB);
  bf16_t* qr   = (bf16_t*)(ws + 40 * MB);    //  8 MB: RoPE'd Q [b][h][s][hd]
  bf16_t* kr   = (bf16_t*)(ws + 48 * MB);    //  8 MB: RoPE'd K [b][h][s][hd]
  bf16_t* vtr  = (bf16_t*)(ws + 56 * MB);    //  8 MB: V transposed [b][h][hd][s]
  bf16_t* attn = (bf16_t*)(ws + 16 * MB);    //  reuse qtmp region (dead after rope)

  const int nx = B_ * S_ * D_;               // 4M elements
  k_cvt_x<<<nx / 256, 256, 0, stream>>>(x, xb, nx);
  k_cvt_w<<<dim3(D_ * D_ / 256, 4), 256, 0, stream>>>(Wq, Wk, Wv, Wout,
                                                      wtq, wtk, wtv, wto);
  k_gemm_qkv<<<dim3(B_ * S_ / 32, D_ / 64, 3), 32, 0, stream>>>(
      xb, wtq, wtk, wtv, qtmp, ktmp, vtmp);
  k_rope_pack<<<(B_ * S_ * H_ * HALF_) / 256, 256, 0, stream>>>(
      qtmp, ktmp, vtmp, fcos, fsin, qr, kr, vtr);
  k_attn<<<dim3(S_ / 16, H_, B_), 32, 0, stream>>>(qr, kr, vtr, attn);
  k_gemm_out<<<dim3(B_ * S_ / 32, D_ / 64), 32, 0, stream>>>(attn, wto, out);
}
